// Swin_multi_21638045237916
// MI455X (gfx1250) — matrix operand order repaired and checked by reading, not compiled, not run on hardware
//
#include <hip/hip_runtime.h>
#include <math.h>

// ---------------- constants ----------------
#define BN    8
#define DIMC  96
#define HH    128
#define WW    128
#define HEADS 16
#define WSZ   4
#define HD    6
#define NT    16          // tokens per window
#define NWIN  8192        // BN * 32 * 32

typedef _Float16 v16h __attribute__((ext_vector_type(16)));
typedef float    v8f  __attribute__((ext_vector_type(8)));
typedef const __attribute__((address_space(1))) _Float16 gf16_t;

// ---------------- WMMA fragment helpers (CDNA5 wave32 layouts) ----------------
// A (16x32 f16): lane&15 = row M; lane>>4 selects K-halves.
__device__ __forceinline__ v16h load_a_frag(const _Float16* base, int strideH, int kt, int lane) {
    const int row  = lane & 15;
    const int half = lane >> 4;
    const _Float16* p = base + row * strideH + kt * 32 + half * 8;
    v16h a;
#pragma unroll
    for (int i = 0; i < 8; ++i) a[i] = p[i];
#pragma unroll
    for (int i = 0; i < 8; ++i) a[8 + i] = p[16 + i];
    return a;
}

// B (32x16 f16) pre-packed: element e = (frag*32 + lane)*16 + i  (global AS -> global_load_b128)
__device__ __forceinline__ v16h load_b_frag(const _Float16* pack, int frag, int lane) {
    gf16_t* p = (gf16_t*)pack + ((size_t)frag * 32 + lane) * 16;
    v16h b;
#pragma unroll
    for (int i = 0; i < 16; ++i) b[i] = p[i];
    return b;
}

__device__ __forceinline__ v8f wmma_f16(v16h a, v16h b, v8f c) {
    return __builtin_amdgcn_wmma_f32_16x16x32_f16(false, a, false, b, (short)0, c, false, false);
}

__device__ __forceinline__ float gelu_exact(float x) {
    return 0.5f * x * (1.0f + erff(x * 0.70710678118654752f));
}

// wid is uniform per wave; force it into an SGPR so task branches become scalar
__device__ __forceinline__ int uni(int v) { return __builtin_amdgcn_readfirstlane(v); }

// ---------------- K0: continuous position bias + inv(clip(tau)) ----------------
__global__ void bias_kernel(const float* __restrict__ w1, const float* __restrict__ b1,
                            const float* __restrict__ w2, const float* __restrict__ b2,
                            const float* __restrict__ tau,
                            float* __restrict__ biasT, float* __restrict__ invTau) {
    const int p = threadIdx.x;            // 0..255 = i*16 + j
    const int i = p >> 4, j = p & 15;
    float d0 = (float)((i >> 2) - (j >> 2));
    float d1 = (float)((i & 3) - (j & 3));
    float l0 = (d0 > 0.f ? 1.f : (d0 < 0.f ? -1.f : 0.f)) * log1pf(fabsf(d0));
    float l1 = (d1 > 0.f ? 1.f : (d1 < 0.f ? -1.f : 0.f)) * log1pf(fabsf(d1));
    float o[HEADS];
#pragma unroll
    for (int h = 0; h < HEADS; ++h) o[h] = b2[h];
    for (int k = 0; k < 256; ++k) {
        float hv = l0 * w1[k] + l1 * w1[256 + k] + b1[k];
        hv = fmaxf(hv, 0.f);
#pragma unroll
        for (int h = 0; h < HEADS; ++h) o[h] += hv * w2[k * HEADS + h];
    }
#pragma unroll
    for (int h = 0; h < HEADS; ++h) {
        biasT[h * 256 + p]  = o[h];
        invTau[h * 256 + p] = 1.0f / fmaxf(tau[(h * 16 + i) * 16 + j], 0.01f);
    }
}

// ---------------- K-pack: f32 weights -> f16 B-fragment order ----------------
__global__ void pack_kernel(const float* __restrict__ src, _Float16* __restrict__ dst,
                            int Ncols, int nKt, int total) {
    int e = blockIdx.x * blockDim.x + threadIdx.x;
    if (e >= total) return;
    int i    = e & 15;
    int lane = (e >> 4) & 31;
    int frag = e >> 9;
    int kt = frag % nKt;
    int nt = frag / nKt;
    int k = kt * 32 + (lane >> 4) * 8 + (i & 7) + (i >> 3) * 16;
    int n = nt * 16 + (lane & 15);
    dst[e] = (_Float16)src[(size_t)k * Ncols + n];
}

// ---------------- K1: fused window-attention block ----------------
// LDS regions (bytes)
#define OFF_TOKS   0        // 3*16*96 f16 = 9216    (aliased by projin 3072)
#define OFF_Q      9216     // 16*96 f32 = 6144
#define OFF_KX     15360    // 6144                  (aliased by projout)
#define OFF_KA     21504    // 6144
#define OFF_VA     27648    // 6144
#define OFF_KD     33792    // 6144
#define OFF_VD     39936    // 6144
#define OFF_KN     46080    // 256 f32 = 1024
#define OFF_XCIN   47104    // 12*256 f32 = 12288
#define OFF_CW     59392    // 648 f32 = 2592
#define OFF_CB     61984    // 6 f32 = 24
#define OFF_MEAN   62016    // 16 f32
#define OFF_RSTD   62080    // 16 f32
#define SMEM_BYTES 62144

__global__ __launch_bounds__(128) void attn_block_kernel(
    const float* __restrict__ x, const float* __restrict__ s1a, const float* __restrict__ s1d,
    const float* __restrict__ qkv_b, const float* __restrict__ qkva_b, const float* __restrict__ qkvd_b,
    const float* __restrict__ biasT, const float* __restrict__ invTau,
    const _Float16* __restrict__ pk_qkvx, const _Float16* __restrict__ pk_qkva,
    const _Float16* __restrict__ pk_qkvd, const _Float16* __restrict__ pk_proj,
    const float* __restrict__ conv_w, const float* __restrict__ conv_b,
    const float* __restrict__ ln_g, const float* __restrict__ ln_b,
    const float* __restrict__ proj_b, const float* __restrict__ n1g, const float* __restrict__ n1b,
    float* __restrict__ out) {

    __shared__ __align__(16) char smem[SMEM_BYTES];
    _Float16* toks    = (_Float16*)(smem + OFF_TOKS);
    _Float16* projin  = (_Float16*)(smem + OFF_TOKS);   // alias (toks dead)
    float* qld     = (float*)(smem + OFF_Q);
    float* kxld    = (float*)(smem + OFF_KX);
    float* projout = (float*)(smem + OFF_KX);           // alias (kx dead)
    float* kald    = (float*)(smem + OFF_KA);
    float* vald    = (float*)(smem + OFF_VA);
    float* kdld    = (float*)(smem + OFF_KD);
    float* vdld    = (float*)(smem + OFF_VD);
    float* knld    = (float*)(smem + OFF_KN);
    float* xcin    = (float*)(smem + OFF_XCIN);
    float* cw      = (float*)(smem + OFF_CW);
    float* cb      = (float*)(smem + OFF_CB);
    float* meansh  = (float*)(smem + OFF_MEAN);
    float* rstdsh  = (float*)(smem + OFF_RSTD);

    const int tid  = threadIdx.x;
    const int lane = tid & 31;
    const int wid  = uni(tid >> 5);      // wave-uniform, force SGPR

    const int w   = blockIdx.x;
    const int b   = w >> 10;
    const int rem = w & 1023;
    const int h0  = (rem >> 5) * WSZ;
    const int w0  = (rem & 31) * WSZ;

    // ---- phase 1: gather window tokens (f16) + conv weights ----
    const float* srcs[3] = {x, s1a, s1d};
    for (int idx = tid; idx < 3 * NT * DIMC; idx += 128) {
        int s = idx / 1536;
        int r = idx - s * 1536;
        int n = r / DIMC, c = r - n * DIMC;
        int ti = n >> 2, tj = n & 3;
        float v = srcs[s][(((size_t)b * DIMC + c) * HH + (h0 + ti)) * WW + (w0 + tj)];
        toks[idx] = (_Float16)v;
    }
    for (int idx = tid; idx < 648; idx += 128) cw[idx] = conv_w[idx];
    if (tid < 6) cb[tid] = conv_b[tid];
    __syncthreads();

    // ---- phase 2: QKV WMMA (only q/k of x, k/v of a, k/v of d) ----
    // task indices are scalar (wid is uniform) -> scalar branches, no EXEC masks
    for (int it = 0; it < 9; ++it) {
        const int T = uni(it * 4 + wid);
        int s, nt;
        if (T < 12)      { s = 0; nt = T; }
        else if (T < 24) { s = 1; nt = T - 6; }
        else             { s = 2; nt = T - 18; }

        const _Float16* pk;
        const float*    bias;
        if (s == 0)      { pk = pk_qkvx; bias = qkv_b;  }
        else if (s == 1) { pk = pk_qkva; bias = qkva_b; }
        else             { pk = pk_qkvd; bias = qkvd_b; }

        v8f c = {0.f, 0.f, 0.f, 0.f, 0.f, 0.f, 0.f, 0.f};
        const _Float16* aBase = toks + s * 1536;
#pragma unroll
        for (int kt = 0; kt < 3; ++kt) {
            v16h a = load_a_frag(aBase, DIMC, kt, lane);
            v16h bf = load_b_frag(pk, nt * 3 + kt, lane);
            c = wmma_f16(a, bf, c);
        }

        // scalar destination select (tile boundaries are multiples of 16)
        float* dst;
        int    cb0;
        float  sv = 1.0f;
        if (s == 0) {
            if (nt < 6) { dst = qld;  cb0 = nt * 16;       sv = 0.40824829046386302f; }
            else        { dst = kxld; cb0 = nt * 16 - 96;  }
        } else {
            float* kdst = (s == 1) ? kald : kdld;
            float* vdst = (s == 1) ? vald : vdld;
            if (nt < 12) { dst = kdst; cb0 = nt * 16 - 96;  }
            else         { dst = vdst; cb0 = nt * 16 - 192; }
        }
        const int   colL  = cb0 + (lane & 15);
        const int   rbase = (lane >> 4) * 8;
        const float bv    = bias[nt * 16 + (lane & 15)];
#pragma unroll
        for (int r = 0; r < 8; ++r)
            dst[(rbase + r) * 96 + colL] = (c[r] + bv) * sv;
    }
    __syncthreads();

    // ---- phase 3: k-norms of SELF (x-stream) k ----
    for (int v = tid; v < 256; v += 128) {
        int h = v >> 4, n = v & 15;
        float s = 0.f;
#pragma unroll
        for (int hd = 0; hd < HD; ++hd) { float t = kxld[n * 96 + h * HD + hd]; s += t * t; }
        knld[v] = sqrtf(s);
    }
    __syncthreads();

    // ---- phase 4: attention, lane = (branch = lane>>4, query i = lane&15) ----
    {
        const int half = lane >> 4;
        const int i    = lane & 15;
        const float* kb = half ? kdld : kald;
        const float* vb = half ? vdld : vald;
        for (int hh2 = 0; hh2 < 4; ++hh2) {
            const int h = uni(wid * 4 + hh2);
            float qv[HD], qn2 = 0.f;
#pragma unroll
            for (int hd = 0; hd < HD; ++hd) { qv[hd] = qld[i * 96 + h * HD + hd]; qn2 += qv[hd] * qv[hd]; }
            float qn = sqrtf(qn2);
            const float* bpt = biasT  + h * 256 + i * 16;
            const float* tpt = invTau + h * 256 + i * 16;
            float sc[NT], mx = -1e30f;
#pragma unroll
            for (int j = 0; j < NT; ++j) {
                float d = 0.f;
#pragma unroll
                for (int hd = 0; hd < HD; ++hd) d += qv[hd] * kb[j * 96 + h * HD + hd];
                float dn = fmaxf(qn * knld[h * 16 + j], 1e-6f);
                float a = d / dn * tpt[j] + bpt[j];
                sc[j] = a;
                mx = fmaxf(mx, a);
            }
            float sum = 0.f;
#pragma unroll
            for (int j = 0; j < NT; ++j) { float e = __expf(sc[j] - mx); sc[j] = e; sum += e; }
            float inv = 1.f / sum;
#pragma unroll
            for (int hd = 0; hd < HD; ++hd) {
                float o = 0.f;
#pragma unroll
                for (int j = 0; j < NT; ++j) o += sc[j] * vb[j * 96 + h * HD + hd];
                xcin[(half * HD + hd) * 256 + h * 16 + i] = o * inv;
            }
        }
    }
    __syncthreads();

    // ---- phase 5: 3x3 conv (spatial = heads x tokens) + LN(HD) + GELU ----
    for (int site = tid; site < 256; site += 128) {
        int y = site >> 4, xt = site & 15;
        float acc[HD];
#pragma unroll
        for (int oc = 0; oc < HD; ++oc) acc[oc] = cb[oc];
        for (int ic = 0; ic < 12; ++ic) {
#pragma unroll
            for (int dy = 0; dy < 3; ++dy) {
                int yy = y + dy - 1;
                if ((unsigned)yy >= 16u) continue;
#pragma unroll
                for (int dx = 0; dx < 3; ++dx) {
                    int xx = xt + dx - 1;
                    if ((unsigned)xx >= 16u) continue;
                    float v = xcin[ic * 256 + yy * 16 + xx];
#pragma unroll
                    for (int oc = 0; oc < HD; ++oc)
                        acc[oc] += cw[((oc * 12 + ic) * 3 + dy) * 3 + dx] * v;
                }
            }
        }
        float m = 0.f;
#pragma unroll
        for (int oc = 0; oc < HD; ++oc) m += acc[oc];
        m *= (1.f / 6.f);
        float var = 0.f;
#pragma unroll
        for (int oc = 0; oc < HD; ++oc) { float d = acc[oc] - m; var += d * d; }
        var *= (1.f / 6.f);
        float rs = rsqrtf(var + 1e-5f);
#pragma unroll
        for (int oc = 0; oc < HD; ++oc) {
            float v = (acc[oc] - m) * rs * ln_g[oc] + ln_b[oc];
            projin[xt * 96 + y * HD + oc] = (_Float16)gelu_exact(v);
        }
    }
    __syncthreads();

    // ---- phase 6: proj WMMA (16x96 @ 96x96); tiles 0..5, waves 0/1 take two ----
    for (int it = 0; it < 2; ++it) {
        const int nt = uni(it * 4 + wid);
        if (nt >= 6) break;
        v8f c = {0.f, 0.f, 0.f, 0.f, 0.f, 0.f, 0.f, 0.f};
#pragma unroll
        for (int kt = 0; kt < 3; ++kt) {
            v16h a = load_a_frag(projin, 96, kt, lane);
            v16h bf = load_b_frag(pk_proj, nt * 3 + kt, lane);
            c = wmma_f16(a, bf, c);
        }
        const int col   = nt * 16 + (lane & 15);
        const int rbase = (lane >> 4) * 8;
        const float bv  = proj_b[col];
#pragma unroll
        for (int r = 0; r < 8; ++r) projout[(rbase + r) * 96 + col] = c[r] + bv;
    }
    __syncthreads();

    // ---- phase 7: LN(96) + shortcut residual -> d_out ----
    if (tid < 16) {
        float m = 0.f;
        for (int c = 0; c < 96; ++c) m += projout[tid * 96 + c];
        m *= (1.f / 96.f);
        float var = 0.f;
        for (int c = 0; c < 96; ++c) { float d = projout[tid * 96 + c] - m; var += d * d; }
        var *= (1.f / 96.f);
        meansh[tid] = m;
        rstdsh[tid] = rsqrtf(var + 1e-5f);
    }
    __syncthreads();
    for (int idx = tid; idx < NT * DIMC; idx += 128) {
        int row = idx / 96, c = idx - row * 96;
        int ti = row >> 2, tj = row & 3;
        float v = (projout[idx] - meansh[row]) * rstdsh[row] * n1g[c] + n1b[c];
        float sc = x[(((size_t)b * DIMC + c) * HH + (h0 + ti)) * WW + (w0 + tj)];
        size_t hw = (size_t)(h0 + ti) * WW + (w0 + tj);
        out[((size_t)b * (HH * WW) + hw) * DIMC + c] = v + sc;
    }
}

// ---------------- K2: fused MLP (in-place on d_out, 16-row tiles) ----------------
__global__ __launch_bounds__(128) void mlp_kernel(
    const _Float16* __restrict__ pk_fc1, const _Float16* __restrict__ pk_fc2,
    const float* __restrict__ fc1_b, const float* __restrict__ fc2_b,
    const float* __restrict__ n2g, const float* __restrict__ n2b,
    float* __restrict__ out) {

    __shared__ __align__(16) _Float16 yh[16 * 96];
    __shared__ __align__(16) float    y32[16 * 96];
    __shared__ __align__(16) _Float16 hh[16 * 384];
    __shared__ __align__(16) float    m32[16 * 96];
    __shared__ float meansh[16], rstdsh[16];

    const int tid  = threadIdx.x;
    const int lane = tid & 31;
    const int wid  = uni(tid >> 5);
    const size_t base = (size_t)blockIdx.x * 16 * 96;

    for (int idx = tid; idx < 1536; idx += 128) {
        float v = out[base + idx];
        y32[idx] = v;
        yh[idx]  = (_Float16)v;
    }
    __syncthreads();

    // fc1 + GELU : 16x96 @ 96x384
    for (int it = 0; it < 6; ++it) {
        const int nt = uni(it * 4 + wid);
        v8f c = {0.f, 0.f, 0.f, 0.f, 0.f, 0.f, 0.f, 0.f};
#pragma unroll
        for (int kt = 0; kt < 3; ++kt) {
            v16h a = load_a_frag(yh, 96, kt, lane);
            v16h bf = load_b_frag(pk_fc1, nt * 3 + kt, lane);
            c = wmma_f16(a, bf, c);
        }
        const int col   = nt * 16 + (lane & 15);
        const int rbase = (lane >> 4) * 8;
        const float bv  = fc1_b[col];
#pragma unroll
        for (int r = 0; r < 8; ++r)
            hh[(rbase + r) * 384 + col] = (_Float16)gelu_exact(c[r] + bv);
    }
    __syncthreads();

    // fc2 : 16x384 @ 384x96; tiles 0..5, waves 0/1 take two
    for (int it = 0; it < 2; ++it) {
        const int nt = uni(it * 4 + wid);
        if (nt >= 6) break;
        v8f c = {0.f, 0.f, 0.f, 0.f, 0.f, 0.f, 0.f, 0.f};
#pragma unroll
        for (int kt = 0; kt < 12; ++kt) {
            v16h a = load_a_frag(hh, 384, kt, lane);
            v16h bf = load_b_frag(pk_fc2, nt * 12 + kt, lane);
            c = wmma_f16(a, bf, c);
        }
        const int col   = nt * 16 + (lane & 15);
        const int rbase = (lane >> 4) * 8;
        const float bv  = fc2_b[col];
#pragma unroll
        for (int r = 0; r < 8; ++r) m32[(rbase + r) * 96 + col] = c[r] + bv;
    }
    __syncthreads();

    if (tid < 16) {
        float m = 0.f;
        for (int c = 0; c < 96; ++c) m += m32[tid * 96 + c];
        m *= (1.f / 96.f);
        float var = 0.f;
        for (int c = 0; c < 96; ++c) { float d = m32[tid * 96 + c] - m; var += d * d; }
        var *= (1.f / 96.f);
        meansh[tid] = m;
        rstdsh[tid] = rsqrtf(var + 1e-5f);
    }
    __syncthreads();
    for (int idx = tid; idx < 1536; idx += 128) {
        int row = idx / 96, c = idx - row * 96;
        out[base + idx] = y32[idx] + (m32[idx] - meansh[row]) * rstdsh[row] * n2g[c] + n2b[c];
    }
}

// ---------------- host ----------------
extern "C" void kernel_launch(void* const* d_in, const int* in_sizes, int n_in,
                              void* d_out, int out_size, void* d_ws, size_t ws_size,
                              hipStream_t stream) {
    (void)in_sizes; (void)n_in; (void)out_size; (void)ws_size;

    const float* x      = (const float*)d_in[0];
    const float* s1a    = (const float*)d_in[1];
    const float* s1d    = (const float*)d_in[2];
    const float* qkv_w  = (const float*)d_in[3];
    const float* qkv_b  = (const float*)d_in[4];
    const float* qkva_w = (const float*)d_in[5];
    const float* qkva_b = (const float*)d_in[6];
    const float* qkvd_w = (const float*)d_in[7];
    const float* qkvd_b = (const float*)d_in[8];
    const float* cpb_w1 = (const float*)d_in[9];
    const float* cpb_b1 = (const float*)d_in[10];
    const float* cpb_w2 = (const float*)d_in[11];
    const float* cpb_b2 = (const float*)d_in[12];
    const float* tau    = (const float*)d_in[13];
    const float* conv_w = (const float*)d_in[14];
    const float* conv_b = (const float*)d_in[15];
    const float* ln_g   = (const float*)d_in[16];
    const float* ln_b   = (const float*)d_in[17];
    const float* proj_w = (const float*)d_in[18];
    const float* proj_b = (const float*)d_in[19];
    const float* n1g    = (const float*)d_in[20];
    const float* n1b    = (const float*)d_in[21];
    const float* n2g    = (const float*)d_in[22];
    const float* n2b    = (const float*)d_in[23];
    const float* fc1_w  = (const float*)d_in[24];
    const float* fc1_b  = (const float*)d_in[25];
    const float* fc2_w  = (const float*)d_in[26];
    const float* fc2_b  = (const float*)d_in[27];
    float* out = (float*)d_out;

    // workspace layout
    float*    biasT  = (float*)d_ws;                 // 4096 f32
    float*    invTau = biasT + 4096;                 // 4096 f32
    _Float16* packB  = (_Float16*)(invTau + 4096);
    _Float16* pk_qkvx = packB;                       // 3*18*512 = 27648
    _Float16* pk_qkva = packB + 27648;
    _Float16* pk_qkvd = packB + 55296;
    _Float16* pk_proj = packB + 82944;               // 3*6*512  = 9216
    _Float16* pk_fc1  = packB + 92160;               // 3*24*512 = 36864
    _Float16* pk_fc2  = packB + 129024;              // 12*6*512 = 36864

    bias_kernel<<<1, 256, 0, stream>>>(cpb_w1, cpb_b1, cpb_w2, cpb_b2, tau, biasT, invTau);

    pack_kernel<<<(27648 + 255) / 256, 256, 0, stream>>>(qkv_w,  pk_qkvx, 288,  3, 27648);
    pack_kernel<<<(27648 + 255) / 256, 256, 0, stream>>>(qkva_w, pk_qkva, 288,  3, 27648);
    pack_kernel<<<(27648 + 255) / 256, 256, 0, stream>>>(qkvd_w, pk_qkvd, 288,  3, 27648);
    pack_kernel<<<(9216  + 255) / 256, 256, 0, stream>>>(proj_w, pk_proj, 96,   3, 9216);
    pack_kernel<<<(36864 + 255) / 256, 256, 0, stream>>>(fc1_w,  pk_fc1,  384,  3, 36864);
    pack_kernel<<<(36864 + 255) / 256, 256, 0, stream>>>(fc2_w,  pk_fc2,  96,  12, 36864);

    attn_block_kernel<<<NWIN, 128, 0, stream>>>(
        x, s1a, s1d, qkv_b, qkva_b, qkvd_b, biasT, invTau,
        pk_qkvx, pk_qkva, pk_qkvd, pk_proj,
        conv_w, conv_b, ln_g, ln_b, proj_b, n1g, n1b, out);

    mlp_kernel<<<NWIN, 128, 0, stream>>>(pk_fc1, pk_fc2, fc1_b, fc2_b, n2g, n2b, out);
}
